// GCN_DECONF_48704929137156
// MI455X (gfx1250) — matrix operand order repaired and checked by reading, not compiled, not run on hardware
//
#include <hip/hip_runtime.h>
#include <hip/hip_fp16.h>
#include <cstdint>

typedef __attribute__((ext_vector_type(16))) _Float16 v16h;
typedef __attribute__((ext_vector_type(8)))  _Float16 v8h;
typedef __attribute__((ext_vector_type(4)))  _Float16 v4h;
typedef __attribute__((ext_vector_type(8)))  float    v8f;

#define HFEAT 128
#define LDSW  136   // LDS row pitch in halves: 272B rows -> 4-bank shift/lane, conflict-free b128 reads

// ---------------- graph preprocessing ----------------

__global__ void k_fill1(float* __restrict__ p, int n) {
  int i = blockIdx.x * blockDim.x + threadIdx.x;
  if (i < n) p[i] = 1.0f;   // self-loop contributes 1 to every degree
}

__global__ void k_deg_edges(const int* __restrict__ dst, float* __restrict__ deg, int e) {
  int i = blockIdx.x * blockDim.x + threadIdx.x;
  if (i < e) unsafeAtomicAdd(&deg[dst[i]], 1.0f);
}

__global__ void k_dinv(float* __restrict__ deg, int n) {
  int i = blockIdx.x * blockDim.x + threadIdx.x;
  if (i < n) deg[i] = 1.0f / sqrtf(deg[i]);
}

__global__ void k_norm(const int* __restrict__ src, const int* __restrict__ dst,
                       const float* __restrict__ dinv, float* __restrict__ nrm, int e) {
  int i = blockIdx.x * blockDim.x + threadIdx.x;
  if (i < e) nrm[i] = dinv[src[i]] * dinv[dst[i]];
}

__global__ void k_f32_to_f16(const float* __restrict__ in, _Float16* __restrict__ out, int n4) {
  int i = blockIdx.x * blockDim.x + threadIdx.x;
  if (i < n4) {
    float4 v = ((const float4*)in)[i];
    v4h h;
    h[0] = (_Float16)v.x; h[1] = (_Float16)v.y; h[2] = (_Float16)v.z; h[3] = (_Float16)v.w;
    ((v4h*)out)[i] = h;
  }
}

// ---------------- WMMA GEMM: C[n,128] = A[n,128](f16) @ W[128,128](f32) ----------------
// mode 0: plain f32 store (layer linear, bias added after aggregation)
// mode 1: fused bias + ReLU (head layers)
__global__ __launch_bounds__(256) void k_gemm_wmma(
    const _Float16* __restrict__ A, const float* __restrict__ W,
    const float* __restrict__ bias, float* __restrict__ C, int n, int mode) {
  __shared__ _Float16 sW[HFEAT * LDSW];  // W transposed: sW[col*LDSW + k]
  __shared__ float    sB[HFEAT];

  int tid = threadIdx.x;
  for (int idx = tid; idx < HFEAT * HFEAT; idx += 256) {
    int k = idx >> 7, c = idx & 127;
    sW[c * LDSW + k] = (_Float16)W[idx];          // f32 -> f16 convert + transpose
  }
  if (tid < HFEAT) sB[tid] = (mode == 1) ? bias[tid] : 0.0f;
  __syncthreads();

  int wave = tid >> 5, lane = tid & 31;
  int rowBase = blockIdx.x * 128 + wave * 16;     // 16-row strip per wave
  if (rowBase >= n) return;
  int rl = lane & 15, kh = lane >> 4;             // row-in-tile, K-half select
  int r = rowBase + rl; if (r >= n) r = n - 1;    // clamp (stores are guarded)
  const _Float16* aRow = A + (size_t)r * HFEAT;

  v8f acc[8];
  #pragma unroll
  for (int ct = 0; ct < 8; ++ct)
    #pragma unroll
    for (int j = 0; j < 8; ++j) acc[ct][j] = 0.0f;

  #pragma unroll
  for (int kc = 0; kc < 4; ++kc) {                // K = 128 in 4 steps of 32
    // A 16x32 f16 fragment: lane holds K = kc*32 + {8*kh..8*kh+7, 16+8*kh..16+8*kh+7}
    v8h a0 = *(const v8h*)(aRow + kc * 32 + kh * 8);
    v8h a1 = *(const v8h*)(aRow + kc * 32 + 16 + kh * 8);
    v16h af;
    #pragma unroll
    for (int i = 0; i < 8; ++i) { af[i] = a0[i]; af[i + 8] = a1[i]; }
    #pragma unroll
    for (int ct = 0; ct < 8; ++ct) {              // all 8 column tiles of H=128
      // B 32x16 f16 fragment: lane holds col = ct*16+rl, K = kc*32 + {8*kh..8*kh+7, 16+8*kh..16+8*kh+7}
      const _Float16* bp = &sW[(ct * 16 + rl) * LDSW + kc * 32 + kh * 8];
      v8h b0 = *(const v8h*)bp;
      v8h b1 = *(const v8h*)(bp + 16);
      v16h bf;
      #pragma unroll
      for (int i = 0; i < 8; ++i) { bf[i] = b0[i]; bf[i + 8] = b1[i]; }
      acc[ct] = __builtin_amdgcn_wmma_f32_16x16x32_f16(
          false, af, false, bf, (short)0, acc[ct], false, false);
    }
  }

  // C layout: VGPR j holds row rowBase + 8*kh + j, col = ct*16 + rl
  #pragma unroll
  for (int ct = 0; ct < 8; ++ct) {
    int col = ct * 16 + rl;
    #pragma unroll
    for (int j = 0; j < 8; ++j) {
      int row = rowBase + kh * 8 + j;
      if (row < n) {
        float v = acc[ct][j];
        if (mode == 1) v = fmaxf(v + sB[col], 0.0f);
        C[(size_t)row * HFEAT + col] = v;
      }
    }
  }
}

// ---------------- aggregation ----------------

// OUT[i] = XW[i] * dinv[i]^2  (self-loop term); thread per (node, 4 features)
__global__ void k_agg_init(const float* __restrict__ xw, const float* __restrict__ dinv,
                           float* __restrict__ out, int n) {
  int idx = blockIdx.x * blockDim.x + threadIdx.x;
  int node = idx >> 5, q = idx & 31;
  if (node >= n) return;
  float s = dinv[node]; s *= s;
  float4 v = ((const float4*)(xw + (size_t)node * HFEAT))[q];
  v.x *= s; v.y *= s; v.z *= s; v.w *= s;
  ((float4*)(out + (size_t)node * HFEAT))[q] = v;
}

// warp per edge: coalesced 512B gather of XW[src], scaled fp32 atomic scatter to OUT[dst]
__global__ void k_edge_agg(const float* __restrict__ xw, float* __restrict__ out,
                           const int* __restrict__ src, const int* __restrict__ dst,
                           const float* __restrict__ nrm, int e) {
  int warp = (blockIdx.x * blockDim.x + threadIdx.x) >> 5;
  int lane = threadIdx.x & 31;
  if (warp >= e) return;
  int s = src[warp], d = dst[warp];
  float nm = nrm[warp];
  float4 v = ((const float4*)(xw + (size_t)s * HFEAT))[lane];
  float* o = out + (size_t)d * HFEAT + lane * 4;
  unsafeAtomicAdd(o + 0, v.x * nm);
  unsafeAtomicAdd(o + 1, v.y * nm);
  unsafeAtomicAdd(o + 2, v.z * nm);
  unsafeAtomicAdd(o + 3, v.w * nm);
}

// in-place bias + ReLU on fp32 rep, plus f16 re-encode for next GEMM
__global__ void k_bias_relu_f16(float* __restrict__ acc, const float* __restrict__ bias,
                                _Float16* __restrict__ f16, int n) {
  int idx = blockIdx.x * blockDim.x + threadIdx.x;
  int node = idx >> 5, q = idx & 31;
  if (node >= n) return;
  float4 b = ((const float4*)bias)[q];
  float4 v = ((float4*)(acc + (size_t)node * HFEAT))[q];
  v.x = fmaxf(v.x + b.x, 0.0f); v.y = fmaxf(v.y + b.y, 0.0f);
  v.z = fmaxf(v.z + b.z, 0.0f); v.w = fmaxf(v.w + b.w, 0.0f);
  ((float4*)(acc + (size_t)node * HFEAT))[q] = v;
  v4h h;
  h[0] = (_Float16)v.x; h[1] = (_Float16)v.y; h[2] = (_Float16)v.z; h[3] = (_Float16)v.w;
  ((v4h*)(f16 + (size_t)node * HFEAT))[q] = h;
}

// ---------------- heads: warp-per-node 128->1 projections ----------------

__global__ void k_head_select(const float* __restrict__ feat, const float* __restrict__ w,
                              const float* __restrict__ b, const int* __restrict__ t,
                              int want_pos, float* __restrict__ out, int n) {
  int node = (blockIdx.x * blockDim.x + threadIdx.x) >> 5;
  int lane = threadIdx.x & 31;
  if (node >= n) return;
  if ((t[node] > 0) != (want_pos != 0)) return;   // warp-uniform branch
  float4 f = ((const float4*)(feat + (size_t)node * HFEAT))[lane];
  float4 ww = ((const float4*)w)[lane];
  float v = f.x * ww.x + f.y * ww.y + f.z * ww.z + f.w * ww.w;
  #pragma unroll
  for (int o = 16; o > 0; o >>= 1) v += __shfl_xor(v, o, 32);
  if (lane == 0) out[node] = v + b[0];
}

__global__ void k_head_sigmoid(const float* __restrict__ feat, const float* __restrict__ w,
                               const float* __restrict__ b, float* __restrict__ out, int n) {
  int node = (blockIdx.x * blockDim.x + threadIdx.x) >> 5;
  int lane = threadIdx.x & 31;
  if (node >= n) return;
  float4 f = ((const float4*)(feat + (size_t)node * HFEAT))[lane];
  float4 ww = ((const float4*)w)[lane];
  float v = f.x * ww.x + f.y * ww.y + f.z * ww.z + f.w * ww.w;
  #pragma unroll
  for (int o = 16; o > 0; o >>= 1) v += __shfl_xor(v, o, 32);
  if (lane == 0) out[node] = 1.0f / (1.0f + expf(-(v + b[0])));
}

// ---------------- orchestration ----------------

extern "C" void kernel_launch(void* const* d_in, const int* in_sizes, int n_in,
                              void* d_out, int out_size, void* d_ws, size_t ws_size,
                              hipStream_t stream) {
  (void)n_in; (void)out_size; (void)ws_size;
  const float* x   = (const float*)d_in[0];
  const int*   t   = (const int*)d_in[1];
  // d_in[2] (z) unused by the reference
  const int*   eix = (const int*)d_in[3];
  const float* W1  = (const float*)d_in[4];  const float* b1  = (const float*)d_in[5];
  const float* W2  = (const float*)d_in[6];  const float* b2  = (const float*)d_in[7];
  const float* W00 = (const float*)d_in[8];  const float* b00 = (const float*)d_in[9];
  const float* W10 = (const float*)d_in[10]; const float* b10 = (const float*)d_in[11];
  const float* W01 = (const float*)d_in[12]; const float* b01 = (const float*)d_in[13];
  const float* W11 = (const float*)d_in[14]; const float* b11 = (const float*)d_in[15];
  const float* Wpp = (const float*)d_in[16]; const float* bpp = (const float*)d_in[17];

  const int N = in_sizes[1];       // t is [N]
  const int E = in_sizes[3] / 2;   // edge_index is [2,E]
  const int* srcI = eix;
  const int* dstI = eix + E;

  // workspace layout (16B aligned)
  char* wsb = (char*)d_ws;
  size_t off = 0;
  auto a16 = [](size_t v) { return (v + 15) & ~(size_t)15; };
  float*    dinv = (float*)(wsb + off);    off = a16(off + (size_t)N * 4);
  float*    nrm  = (float*)(wsb + off);    off = a16(off + (size_t)E * 4);
  float*    XW   = (float*)(wsb + off);    off = a16(off + (size_t)N * HFEAT * 4);
  float*    REP  = (float*)(wsb + off);    off = a16(off + (size_t)N * HFEAT * 4);
  _Float16* F16  = (_Float16*)(wsb + off);

  float* pOut = (float*)d_out;     // p1 in [0,N)
  float* yOut = pOut + N;          // y  in [N,2N)

  const int thr = 256;
  const int gN     = (N + thr - 1) / thr;
  const int gE     = (E + thr - 1) / thr;
  const int gN4    = (N * (HFEAT / 4) + thr - 1) / thr;     // node*32 threads
  const int gNode  = (N * 32 + thr - 1) / thr;              // warp-per-node / per (node,4feat)
  const int gEdgeW = (int)(((long long)E * 32 + thr - 1) / thr);  // warp-per-edge
  const int gGemm  = (N + 127) / 128;

  // degrees -> dinv -> per-edge norm
  k_fill1    <<<gN, thr, 0, stream>>>(dinv, N);
  k_deg_edges<<<gE, thr, 0, stream>>>(dstI, dinv, E);
  k_dinv     <<<gN, thr, 0, stream>>>(dinv, N);
  k_norm     <<<gE, thr, 0, stream>>>(srcI, dstI, dinv, nrm, E);

  // f16 copy of input features
  k_f32_to_f16<<<gN4, thr, 0, stream>>>(x, F16, N * (HFEAT / 4));

  // ---- GCN layer 1 ----
  k_gemm_wmma    <<<gGemm, 256, 0, stream>>>(F16, W1, b1, XW, N, 0);
  k_agg_init     <<<gNode, thr, 0, stream>>>(XW, dinv, REP, N);
  k_edge_agg     <<<gEdgeW, thr, 0, stream>>>(XW, REP, srcI, dstI, nrm, E);
  k_bias_relu_f16<<<gNode, thr, 0, stream>>>(REP, b1, F16, N);

  // ---- GCN layer 2 ----
  k_gemm_wmma    <<<gGemm, 256, 0, stream>>>(F16, W2, b2, XW, N, 0);
  k_agg_init     <<<gNode, thr, 0, stream>>>(XW, dinv, REP, N);
  k_edge_agg     <<<gEdgeW, thr, 0, stream>>>(XW, REP, srcI, dstI, nrm, E);
  k_bias_relu_f16<<<gNode, thr, 0, stream>>>(REP, b2, F16, N);   // REP = final rep (fp32), F16 = f16 rep

  // ---- outcome heads: y = t>0 ? relu(rep@W10+b10)@W11+b11 : relu(rep@W00+b00)@W01+b01 ----
  k_gemm_wmma  <<<gGemm, 256, 0, stream>>>(F16, W00, b00, XW, N, 1);
  k_head_select<<<gNode, thr, 0, stream>>>(XW, W01, b01, t, 0, yOut, N);
  k_gemm_wmma  <<<gGemm, 256, 0, stream>>>(F16, W10, b10, XW, N, 1);
  k_head_select<<<gNode, thr, 0, stream>>>(XW, W11, b11, t, 1, yOut, N);

  // ---- propensity head: p1 = sigmoid(rep@Wpp + bpp) ----
  k_head_sigmoid<<<gNode, thr, 0, stream>>>(REP, Wpp, bpp, pOut, N);
}
